// CausalSelfAttention_48850958025142
// MI455X (gfx1250) — compile-verified, matrix operand order checked by reading, not run on hardware
//
#include <hip/hip_runtime.h>
#include <hip/hip_bf16.h>

typedef unsigned short u16;
typedef __attribute__((ext_vector_type(16))) __bf16 v16bf;
typedef __attribute__((ext_vector_type(8)))  float  v8f;
typedef __attribute__((ext_vector_type(4)))  unsigned int u32x4;

union FragB {
    v16bf v;
    u32x4 q[2];
    u16   h[16];
};

static __device__ __forceinline__ u16 f2bfu(float f) {
    union { float f; unsigned u; } c; c.f = f;
    unsigned u = c.u + 0x7fffu + ((c.u >> 16) & 1u);
    return (u16)(u >> 16);
}

static __device__ __forceinline__ v8f v8zero() {
    v8f z;
#pragma unroll
    for (int i = 0; i < 8; ++i) z[i] = 0.0f;
    return z;
}

// --- gfx1250 async global->LDS copy (ASYNCcnt-tracked), via inline asm ------
typedef __attribute__((address_space(3))) u16 lds_u16;

static __device__ __forceinline__ unsigned lds_off_of(u16* p) {
    // addrspace(3) pointers are 32-bit offsets from LDS_BASE
    return (unsigned)(unsigned long long)(lds_u16*)p;
}

static __device__ __forceinline__ void async_load_b128(unsigned lds_off, const void* gaddr) {
    asm volatile("global_load_async_to_lds_b128 %0, %1, off"
                 :: "v"(lds_off), "v"(gaddr)
                 : "memory");
}

static __device__ __forceinline__ void wait_async0() {
    asm volatile("s_wait_asynccnt 0" ::: "memory");
}

// ---------------------------------------------------------------------------
// Kernel 1a: fp32 -> bf16 conversion (plain)
// ---------------------------------------------------------------------------
__global__ void cvt_bf16_kernel(const float* __restrict__ in, u16* __restrict__ out, int n) {
    int i = blockIdx.x * 256 + threadIdx.x;
    if (i < n) out[i] = f2bfu(in[i]);
}

// ---------------------------------------------------------------------------
// Kernel 1b: fp32 -> bf16 conversion with transpose:
//   in  : [rows][cols] fp32   ->   out : [cols][rows] bf16
// (coalesced reads; one-time cost so scattered 2B writes are fine)
// ---------------------------------------------------------------------------
__global__ void cvt_bf16_t_kernel(const float* __restrict__ in, u16* __restrict__ out,
                                  int rows, int cols) {
    int i = blockIdx.x * 256 + threadIdx.x;
    if (i < rows * cols) {
        int k = i / cols;           // source row
        int n = i - k * cols;       // source col
        out[(size_t)n * rows + k] = f2bfu(in[i]);
    }
}

// ---------------------------------------------------------------------------
// Kernel 2: QKV GEMM  (4096 x 3072) = xb(4096x1024) @ Wt(3072x1024)^T
// Both operands k-contiguous; tiles staged with async global->LDS b128 copies.
// Epilogue scatters into Q[bh][t][dh] (scaled 1/8), K[bh][t][dh], Vt[bh][dh][t]
// ---------------------------------------------------------------------------
__global__ __launch_bounds__(256) void qkv_gemm_kernel(
    const u16* __restrict__ xb, const u16* __restrict__ wt,
    u16* __restrict__ q, u16* __restrict__ k, u16* __restrict__ vt) {
    __shared__ __align__(16) u16 As[128 * 40];
    __shared__ __align__(16) u16 Bs[128 * 40];

    const int tid  = threadIdx.x;
    const int lane = tid & 31;
    const int wave = tid >> 5;
    const int l15  = lane & 15;
    const int hi   = lane >> 4;           // 0 or 1
    const int row0 = blockIdx.x * 128;
    const int n0b  = blockIdx.y * 128;
    const int rg   = wave >> 1;           // 0..3 : 32-row group
    const int cg   = wave & 1;            // 0..1 : 64-col group

    const unsigned asb = lds_off_of(As);
    const unsigned bsb = lds_off_of(Bs);

    v8f acc[2][4];
#pragma unroll
    for (int i = 0; i < 2; ++i)
#pragma unroll
        for (int j = 0; j < 4; ++j) acc[i][j] = v8zero();

    for (int k0 = 0; k0 < 1024; k0 += 32) {
        // Stage A (128x32) and Bt (128x32) tiles: async b128 copies to LDS
#pragma unroll
        for (int i = 0; i < 2; ++i) {
            int idx = tid + i * 256;          // 0..511
            int r = idx >> 2, c = idx & 3;
            unsigned loff = (unsigned)(r * 40 + c * 8) * 2u;
            async_load_b128(asb + loff, &xb[(size_t)(row0 + r) * 1024 + k0 + c * 8]);
            async_load_b128(bsb + loff, &wt[(size_t)(n0b + r) * 1024 + k0 + c * 8]);
        }
        wait_async0();
        __syncthreads();

        FragB af[2], bf[4];
#pragma unroll
        for (int rt = 0; rt < 2; ++rt) {
            const u16* p = &As[(rg * 32 + rt * 16 + l15) * 40];
            af[rt].q[0] = *(const u32x4*)(p + hi * 8);
            af[rt].q[1] = *(const u32x4*)(p + 16 + hi * 8);
        }
#pragma unroll
        for (int ct = 0; ct < 4; ++ct) {
            const u16* p = &Bs[(cg * 64 + ct * 16 + l15) * 40];
            bf[ct].q[0] = *(const u32x4*)(p + hi * 8);
            bf[ct].q[1] = *(const u32x4*)(p + 16 + hi * 8);
        }
#pragma unroll
        for (int rt = 0; rt < 2; ++rt)
#pragma unroll
            for (int ct = 0; ct < 4; ++ct)
                acc[rt][ct] = __builtin_amdgcn_wmma_f32_16x16x32_bf16(
                    false, af[rt].v, false, bf[ct].v, (short)0, acc[rt][ct], false, false);
        __syncthreads();
    }

    // Epilogue: scatter into q (scaled), k, v-transposed
#pragma unroll
    for (int rt = 0; rt < 2; ++rt) {
#pragma unroll
        for (int ct = 0; ct < 4; ++ct) {
            int col   = n0b + cg * 64 + ct * 16 + l15;
            int which = col >> 10;            // 0=q 1=k 2=v
            int dcol  = col & 1023;
            int head  = dcol >> 6;
            int dhi   = dcol & 63;
#pragma unroll
            for (int r = 0; r < 8; ++r) {
                int row = row0 + rg * 32 + rt * 16 + r + hi * 8;
                int bb = row >> 11, tt = row & 2047;
                int bh = bb * 16 + head;
                float val = acc[rt][ct][r];
                if (which == 0)
                    q[((size_t)bh * 2048 + tt) * 64 + dhi] = f2bfu(val * 0.125f);
                else if (which == 1)
                    k[((size_t)bh * 2048 + tt) * 64 + dhi] = f2bfu(val);
                else
                    vt[((size_t)bh * 64 + dhi) * 2048 + tt] = f2bfu(val);
            }
        }
    }
}

// ---------------------------------------------------------------------------
// Kernel 3: causal flash attention.
// grid = (T/64, B*H), 128 threads (4 waves); each wave owns 16 query rows.
// ---------------------------------------------------------------------------
__global__ __launch_bounds__(128) void flash_attn_kernel(
    const u16* __restrict__ q, const u16* __restrict__ k,
    const u16* __restrict__ vt, u16* __restrict__ yb) {
    __shared__ __align__(16) u16 P[4 * 16 * 32];   // per-wave private 16x32 staging

    const int tid  = threadIdx.x;
    const int lane = tid & 31;
    const int wave = tid >> 5;
    const int l15  = lane & 15;
    const int hi   = lane >> 4;
    const int bh   = blockIdx.y;                   // 0..31
    const int q0   = blockIdx.x * 64 + wave * 16;  // first query row of this wave
    const size_t kvBase = (size_t)bh * 2048 * 64;

    // Load Q fragments (scale already folded in)
    FragB qf[2];
    {
        const u16* p = &q[kvBase + (size_t)(q0 + l15) * 64];
#pragma unroll
        for (int f = 0; f < 2; ++f) {
            qf[f].q[0] = *(const u32x4*)(p + f * 32 + hi * 8);
            qf[f].q[1] = *(const u32x4*)(p + f * 32 + 16 + hi * 8);
        }
    }

    v8f oacc[4];
#pragma unroll
    for (int c = 0; c < 4; ++c) oacc[c] = v8zero();
    float mrow[8], lrow[8];
#pragma unroll
    for (int r = 0; r < 8; ++r) { mrow[r] = -3.0e38f; lrow[r] = 0.0f; }

    const int nkb = (q0 + 16 + 31) >> 5;          // 32-key blocks (causal extent)
    u16* Pw = &P[wave * 512];

    for (int kb = 0; kb < nkb; ++kb) {
        const int t0 = kb * 32;

        // S = Q @ K^T for two 16-key tiles
        v8f s[2];
#pragma unroll
        for (int j = 0; j < 2; ++j) {
            FragB kf0, kf1;
            const u16* kp = &k[kvBase + (size_t)(t0 + j * 16 + l15) * 64];
            kf0.q[0] = *(const u32x4*)(kp + hi * 8);
            kf0.q[1] = *(const u32x4*)(kp + 16 + hi * 8);
            kf1.q[0] = *(const u32x4*)(kp + 32 + hi * 8);
            kf1.q[1] = *(const u32x4*)(kp + 48 + hi * 8);
            v8f z = v8zero();
            z = __builtin_amdgcn_wmma_f32_16x16x32_bf16(false, qf[0].v, false, kf0.v, (short)0, z, false, false);
            z = __builtin_amdgcn_wmma_f32_16x16x32_bf16(false, qf[1].v, false, kf1.v, (short)0, z, false, false);
            s[j] = z;
        }

        // Causal mask + online softmax (row stats replicated across 16 lanes)
        float p0[8], p1[8];
#pragma unroll
        for (int r = 0; r < 8; ++r) {
            int row = q0 + r + hi * 8;
            float s0 = ((t0 + l15) <= row)      ? s[0][r] : -3.0e38f;
            float s1 = ((t0 + 16 + l15) <= row) ? s[1][r] : -3.0e38f;
            float mx = fmaxf(s0, s1);
#pragma unroll
            for (int off = 8; off; off >>= 1) mx = fmaxf(mx, __shfl_xor(mx, off, 16));
            float newm = fmaxf(mrow[r], mx);
            float scl  = __expf(mrow[r] - newm);
            float e0   = __expf(s0 - newm);
            float e1   = __expf(s1 - newm);
            float rs   = e0 + e1;
#pragma unroll
            for (int off = 8; off; off >>= 1) rs += __shfl_xor(rs, off, 16);
            lrow[r] = lrow[r] * scl + rs;
            mrow[r] = newm;
#pragma unroll
            for (int c = 0; c < 4; ++c) oacc[c][r] *= scl;
            p0[r] = e0; p1[r] = e1;
        }

        // P: C-layout -> A-layout via per-wave private LDS (DS ops in-order per wave)
#pragma unroll
        for (int r = 0; r < 8; ++r) {
            int rr = r + hi * 8;
            Pw[rr * 32 + l15]      = f2bfu(p0[r]);
            Pw[rr * 32 + 16 + l15] = f2bfu(p1[r]);
        }
        FragB pa;
        {
            const u16* pp = &Pw[l15 * 32];
            pa.q[0] = *(const u32x4*)(pp + hi * 8);
            pa.q[1] = *(const u32x4*)(pp + 16 + hi * 8);
        }

        // O += P @ V  (V stored transposed: [dh][t], k contiguous per lane)
#pragma unroll
        for (int c = 0; c < 4; ++c) {
            FragB vf;
            const u16* vp = &vt[(size_t)bh * 64 * 2048 + (size_t)(c * 16 + l15) * 2048 + t0];
            vf.q[0] = *(const u32x4*)(vp + hi * 8);
            vf.q[1] = *(const u32x4*)(vp + 16 + hi * 8);
            oacc[c] = __builtin_amdgcn_wmma_f32_16x16x32_bf16(
                false, pa.v, false, vf.v, (short)0, oacc[c], false, false);
        }
    }

    // Normalize and store y in (b, t, h, dh) = (b, t, d) layout as bf16
    const int bb = bh >> 4, h = bh & 15;
#pragma unroll
    for (int c = 0; c < 4; ++c) {
#pragma unroll
        for (int r = 0; r < 8; ++r) {
            int row = q0 + r + hi * 8;
            float val = oacc[c][r] / lrow[r];
            yb[(((size_t)bb * 2048 + row) * 16 + h) * 64 + c * 16 + l15] = f2bfu(val);
        }
    }
}

// ---------------------------------------------------------------------------
// Kernel 4: output projection  out(4096x1024) = yb(4096x1024) @ WoutT(1024x1024)^T
// fp32 output; async LDS staging like kernel 2.
// ---------------------------------------------------------------------------
__global__ __launch_bounds__(256) void out_gemm_kernel(
    const u16* __restrict__ yb, const u16* __restrict__ wt, float* __restrict__ out) {
    __shared__ __align__(16) u16 As[128 * 40];
    __shared__ __align__(16) u16 Bs[128 * 40];

    const int tid  = threadIdx.x;
    const int lane = tid & 31;
    const int wave = tid >> 5;
    const int l15  = lane & 15;
    const int hi   = lane >> 4;
    const int row0 = blockIdx.x * 128;
    const int n0b  = blockIdx.y * 128;
    const int rg   = wave >> 1;
    const int cg   = wave & 1;

    const unsigned asb = lds_off_of(As);
    const unsigned bsb = lds_off_of(Bs);

    v8f acc[2][4];
#pragma unroll
    for (int i = 0; i < 2; ++i)
#pragma unroll
        for (int j = 0; j < 4; ++j) acc[i][j] = v8zero();

    for (int k0 = 0; k0 < 1024; k0 += 32) {
#pragma unroll
        for (int i = 0; i < 2; ++i) {
            int idx = tid + i * 256;
            int r = idx >> 2, c = idx & 3;
            unsigned loff = (unsigned)(r * 40 + c * 8) * 2u;
            async_load_b128(asb + loff, &yb[(size_t)(row0 + r) * 1024 + k0 + c * 8]);
            async_load_b128(bsb + loff, &wt[(size_t)(n0b + r) * 1024 + k0 + c * 8]);
        }
        wait_async0();
        __syncthreads();

        FragB af[2], bf[4];
#pragma unroll
        for (int rt = 0; rt < 2; ++rt) {
            const u16* p = &As[(rg * 32 + rt * 16 + l15) * 40];
            af[rt].q[0] = *(const u32x4*)(p + hi * 8);
            af[rt].q[1] = *(const u32x4*)(p + 16 + hi * 8);
        }
#pragma unroll
        for (int ct = 0; ct < 4; ++ct) {
            const u16* p = &Bs[(cg * 64 + ct * 16 + l15) * 40];
            bf[ct].q[0] = *(const u32x4*)(p + hi * 8);
            bf[ct].q[1] = *(const u32x4*)(p + 16 + hi * 8);
        }
#pragma unroll
        for (int rt = 0; rt < 2; ++rt)
#pragma unroll
            for (int ct = 0; ct < 4; ++ct)
                acc[rt][ct] = __builtin_amdgcn_wmma_f32_16x16x32_bf16(
                    false, af[rt].v, false, bf[ct].v, (short)0, acc[rt][ct], false, false);
        __syncthreads();
    }

#pragma unroll
    for (int rt = 0; rt < 2; ++rt) {
#pragma unroll
        for (int ct = 0; ct < 4; ++ct) {
            int col = n0b + cg * 64 + ct * 16 + l15;
#pragma unroll
            for (int r = 0; r < 8; ++r) {
                int row = row0 + rg * 32 + rt * 16 + r + hi * 8;
                out[(size_t)row * 1024 + col] = acc[rt][ct][r];
            }
        }
    }
}

// ---------------------------------------------------------------------------
// Host launcher
// ---------------------------------------------------------------------------
extern "C" void kernel_launch(void* const* d_in, const int* in_sizes, int n_in,
                              void* d_out, int out_size, void* d_ws, size_t ws_size,
                              hipStream_t stream) {
    const float* x     = (const float*)d_in[0];   // (2, 2048, 1024)
    const float* w_qkv = (const float*)d_in[1];   // (1024, 3072)
    const float* w_out = (const float*)d_in[2];   // (1024, 1024)
    float* out = (float*)d_out;                   // (2, 2048, 1024) fp32

    const int XB   = 2 * 2048 * 1024;   // 4194304
    const int WQKV = 1024 * 3072;       // 3145728
    const int WOUT = 1024 * 1024;       // 1048576

    u16* base   = (u16*)d_ws;
    u16* xb     = base;
    u16* wqkvt  = xb + XB;              // transposed: [3072][1024]
    u16* woutt  = wqkvt + WQKV;         // transposed: [1024][1024]
    u16* qb     = woutt + WOUT;
    u16* kb     = qb + XB;
    u16* vtb    = kb + XB;
    u16* ybuf   = vtb + XB;

    cvt_bf16_kernel<<<(XB + 255) / 256, 256, 0, stream>>>(x, xb, XB);
    cvt_bf16_t_kernel<<<(WQKV + 255) / 256, 256, 0, stream>>>(w_qkv, wqkvt, 1024, 3072);
    cvt_bf16_t_kernel<<<(WOUT + 255) / 256, 256, 0, stream>>>(w_out, woutt, 1024, 1024);

    qkv_gemm_kernel<<<dim3(32, 24), 256, 0, stream>>>(xb, wqkvt, qb, kb, vtb);
    flash_attn_kernel<<<dim3(32, 32), 128, 0, stream>>>(qb, kb, vtb, ybuf);
    out_gemm_kernel<<<dim3(32, 8), 256, 0, stream>>>(ybuf, woutt, out);
}
